// StudentHybrid_63685775065403
// MI455X (gfx1250) — matrix operand order repaired and checked by reading, not compiled, not run on hardware
//
#include <hip/hip_runtime.h>
#include <hip/hip_bf16.h>

typedef __attribute__((ext_vector_type(16))) _Float16 v16h;
typedef __attribute__((ext_vector_type(8)))  _Float16 v8h;
typedef __attribute__((ext_vector_type(8)))  float    v8f;

#define LQ    2080      // H0*W0 = 40*52
#define DIM   128
#define NHEAD 4
#define DH    32
#define W0Q   52
#define HFQ   160
#define WFQ   208
#define NPIX  (HFQ*WFQ) // 33280
#define CFQ   32

// ---------------------------------------------------------------------------
// Generic WMMA GEMM: C[M,N] = scale * A[M,K](f32) x B + bias
//   bTrans=1: B is (N,K) row-major  (X @ W^T);  bTrans=0: B is (K,N) row-major
//   biasMode: 0 none, 1 per-column, 2 per-row
// One wave per 32x32 block: 2x2 tiles, 4 WMMAs per K-step of 32.
// ---------------------------------------------------------------------------
__global__ __launch_bounds__(32) void k_gemm(
    const float* __restrict__ A, int lda,
    const float* __restrict__ B, int ldb, int bTrans,
    float* __restrict__ C, int ldc,
    const float* __restrict__ bias, int biasMode, int relu,
    int K, float scale)
{
  const int lane = threadIdx.x & 31;
  const int col0 = blockIdx.x * 32;
  const int row0 = blockIdx.y * 32;

  const int mA  = lane & 15;
  const int khA = (lane >> 4) * 16;
  const int nB  = lane & 15;
  const int khB = (lane >> 4) * 16;

  v8f acc00 = {}, acc01 = {}, acc10 = {}, acc11 = {};

  for (int k0 = 0; k0 < K; k0 += 32) {
    v16h a0, a1, b0, b1;
    const float* pa0 = A + (size_t)(row0 + mA) * lda + k0 + khA;
    const float* pa1 = pa0 + (size_t)16 * lda;
#pragma unroll
    for (int i = 0; i < 8; ++i) {
      a0[i]     = (_Float16)pa0[i];
      a0[8 + i] = (_Float16)pa0[8 + i];
      a1[i]     = (_Float16)pa1[i];
      a1[8 + i] = (_Float16)pa1[8 + i];
    }
    if (bTrans) {
      const float* pb0 = B + (size_t)(col0 + nB) * ldb + k0 + khB;
      const float* pb1 = pb0 + (size_t)16 * ldb;
#pragma unroll
      for (int i = 0; i < 16; ++i) { b0[i] = (_Float16)pb0[i]; b1[i] = (_Float16)pb1[i]; }
    } else {
      const float* pb = B + (size_t)(k0 + khB) * ldb + col0 + nB;
#pragma unroll
      for (int i = 0; i < 16; ++i) {
        b0[i] = (_Float16)pb[(size_t)i * ldb];
        b1[i] = (_Float16)pb[(size_t)i * ldb + 16];
      }
    }
    acc00 = __builtin_amdgcn_wmma_f32_16x16x32_f16(false, a0, false, b0, (short)0, acc00, false, false);
    acc01 = __builtin_amdgcn_wmma_f32_16x16x32_f16(false, a0, false, b1, (short)0, acc01, false, false);
    acc10 = __builtin_amdgcn_wmma_f32_16x16x32_f16(false, a1, false, b0, (short)0, acc10, false, false);
    acc11 = __builtin_amdgcn_wmma_f32_16x16x32_f16(false, a1, false, b1, (short)0, acc11, false, false);
  }

  const int n  = lane & 15;
  const int mb = (lane >> 4) * 8;
  const v8f* accs[4] = { &acc00, &acc01, &acc10, &acc11 };
#pragma unroll
  for (int ti = 0; ti < 2; ++ti) {
#pragma unroll
    for (int tj = 0; tj < 2; ++tj) {
      const v8f& acc = *accs[ti * 2 + tj];
      const int cb = col0 + tj * 16 + n;
      const int rb = row0 + ti * 16 + mb;
      float bcol = (biasMode == 1) ? bias[cb] : 0.f;
#pragma unroll
      for (int v = 0; v < 8; ++v) {
        float r = acc[v] * scale;
        if (biasMode == 1) r += bcol;
        else if (biasMode == 2) r += bias[rb + v];
        if (relu) r = fmaxf(r, 0.f);
        C[(size_t)(rb + v) * ldc + cb] = r;
      }
    }
  }
}

// ---------------------------------------------------------------------------
// Fused flash attention: one wave per (16-row q-tile, head).
// Per 32-column K-strip: 2 score WMMAs -> online softmax (shfl_xor row
// reductions within 16-lane halves; row = vgpr + 8*(lane>=16)) -> P relayout
// C-layout -> A-layout via 1KB LDS bounce -> 2 P@V WMMAs.
// Never materializes the LxL score tensor.
// ---------------------------------------------------------------------------
__global__ __launch_bounds__(32) void k_flash(
    const float* __restrict__ qkv,   // L x 384 (q | k | v)
    float* __restrict__ ohcat)       // L x 128 (head-concat output)
{
  __shared__ __align__(16) _Float16 ph[16 * 32];
  const int lane = threadIdx.x & 31;
  const int row0 = blockIdx.x * 16;
  const int h    = blockIdx.y;
  const int qoff = h * DH, koff = DIM + h * DH, voff = 2 * DIM + h * DH;
  const int half = lane >> 4;
  const int nn   = lane & 15;
  const int khA  = half * 16;
  const int khB  = half * 16;
  const float iscale = 0.17677669529663688f;  // 1/sqrt(32)

  // Q fragment (A-layout, K = dh = 32), loaded once
  v16h qf;
  {
    const float* pq = qkv + (size_t)(row0 + nn) * (3 * DIM) + qoff + khA;
#pragma unroll
    for (int i = 0; i < 8; ++i) { qf[i] = (_Float16)pq[i]; qf[8 + i] = (_Float16)pq[8 + i]; }
  }

  v8f o0 = {}, o1 = {};
  float rm[8], rs[8];
#pragma unroll
  for (int v = 0; v < 8; ++v) { rm[v] = -3.0e38f; rs[v] = 0.f; }

  for (int kt = 0; kt < LQ; kt += 32) {
    // K fragments: B layout from (N=token, K=dh) row-major
    v16h b0, b1;
    const float* pk0 = qkv + (size_t)(kt + nn) * (3 * DIM) + koff + khB;
    const float* pk1 = pk0 + (size_t)16 * (3 * DIM);
#pragma unroll
    for (int i = 0; i < 16; ++i) { b0[i] = (_Float16)pk0[i]; b1[i] = (_Float16)pk1[i]; }

    v8f s0 = {}, s1 = {};
    s0 = __builtin_amdgcn_wmma_f32_16x16x32_f16(false, qf, false, b0, (short)0, s0, false, false);
    s1 = __builtin_amdgcn_wmma_f32_16x16x32_f16(false, qf, false, b1, (short)0, s1, false, false);

    float tm[8];
#pragma unroll
    for (int v = 0; v < 8; ++v) {
      s0[v] *= iscale; s1[v] *= iscale;
      tm[v] = fmaxf(s0[v], s1[v]);
    }
#pragma unroll
    for (int m = 1; m < 16; m <<= 1) {
#pragma unroll
      for (int v = 0; v < 8; ++v) tm[v] = fmaxf(tm[v], __shfl_xor(tm[v], m, 32));
    }

    float ts[8];
#pragma unroll
    for (int v = 0; v < 8; ++v) {
      float nm = fmaxf(rm[v], tm[v]);
      float cf = __expf(rm[v] - nm);
      s0[v] = __expf(s0[v] - nm);
      s1[v] = __expf(s1[v] - nm);
      ts[v] = s0[v] + s1[v];
      rm[v] = nm;
      rs[v] *= cf;
      o0[v] *= cf; o1[v] *= cf;
    }
#pragma unroll
    for (int m = 1; m < 16; m <<= 1) {
#pragma unroll
      for (int v = 0; v < 8; ++v) ts[v] += __shfl_xor(ts[v], m, 32);
    }
#pragma unroll
    for (int v = 0; v < 8; ++v) rs[v] += ts[v];

    // P: C-layout (row = v + 8*half, col = nn [+16]) -> LDS row-major f16
    __syncthreads();
#pragma unroll
    for (int v = 0; v < 8; ++v) {
      ph[(v + 8 * half) * 32 + nn]      = (_Float16)s0[v];
      ph[(v + 8 * half) * 32 + 16 + nn] = (_Float16)s1[v];
    }
    __syncthreads();
    // reread as A-fragment: lane row = nn, K {khA..khA+15} contiguous
    v8h lo = *(const v8h*)&ph[nn * 32 + khA];
    v8h hi = *(const v8h*)&ph[nn * 32 + 8 + khA];
    v16h pf = __builtin_shufflevector(lo, hi, 0, 1, 2, 3, 4, 5, 6, 7,
                                      8, 9, 10, 11, 12, 13, 14, 15);

    // V fragments: B layout from (K=token, N=dh) row-major
    v16h bv0, bv1;
    const float* pv = qkv + (size_t)(kt + khB) * (3 * DIM) + voff + nn;
#pragma unroll
    for (int i = 0; i < 16; ++i) {
      bv0[i] = (_Float16)pv[(size_t)i * (3 * DIM)];
      bv1[i] = (_Float16)pv[(size_t)i * (3 * DIM) + 16];
    }
    o0 = __builtin_amdgcn_wmma_f32_16x16x32_f16(false, pf, false, bv0, (short)0, o0, false, false);
    o1 = __builtin_amdgcn_wmma_f32_16x16x32_f16(false, pf, false, bv1, (short)0, o1, false, false);
  }

  // normalize and store (C layout)
#pragma unroll
  for (int v = 0; v < 8; ++v) {
    float inv = 1.f / rs[v];
    size_t r = (size_t)(row0 + v + 8 * half) * DIM;
    ohcat[r + h * DH + nn]      = o0[v] * inv;
    ohcat[r + h * DH + 16 + nn] = o1[v] * inv;
  }
}

// (C,HW) -> (HW,C) token layout
__global__ void k_tok(const float* __restrict__ c, float* __restrict__ x) {
  int i = blockIdx.x * blockDim.x + threadIdx.x;
  if (i >= LQ * DIM) return;
  int l = i / DIM, d = i % DIM;
  x[i] = c[(size_t)d * LQ + l];
}

// x = LN(x + t) * g + b
__global__ __launch_bounds__(128) void k_add_ln(
    float* __restrict__ x, const float* __restrict__ t,
    const float* __restrict__ g, const float* __restrict__ b)
{
  __shared__ float red[128];
  int l = blockIdx.x, d = threadIdx.x;
  size_t idx = (size_t)l * DIM + d;
  float v = x[idx] + t[idx];
  red[d] = v; __syncthreads();
  for (int s = 64; s > 0; s >>= 1) { if (d < s) red[d] += red[d + s]; __syncthreads(); }
  float m = red[0] * (1.f / DIM); __syncthreads();
  float dv = v - m;
  red[d] = dv * dv; __syncthreads();
  for (int s = 64; s > 0; s >>= 1) { if (d < s) red[d] += red[d + s]; __syncthreads(); }
  float var = red[0] * (1.f / DIM);
  x[idx] = dv * rsqrtf(var + 1e-5f) * g[d] + b[d];
}

__global__ __launch_bounds__(128) void k_l2norm(const float* __restrict__ xin, float* __restrict__ xout) {
  __shared__ float red[128];
  int l = blockIdx.x, d = threadIdx.x;
  float v = xin[(size_t)l * DIM + d];
  red[d] = v * v; __syncthreads();
  for (int s = 64; s > 0; s >>= 1) { if (d < s) red[d] += red[d + s]; __syncthreads(); }
  float inv = 1.f / fmaxf(sqrtf(red[0]), 1e-12f);
  xout[(size_t)l * DIM + d] = v * inv;
}

__global__ __launch_bounds__(256) void k_rowstats(const float* __restrict__ S,
                                                  float* rmax, float* rsum) {
  __shared__ float red[256];
  const float* row = S + (size_t)blockIdx.x * LQ;
  int t = threadIdx.x;
  float mx = -3.4e38f;
  for (int j = t; j < LQ; j += 256) mx = fmaxf(mx, row[j]);
  red[t] = mx; __syncthreads();
  for (int s = 128; s > 0; s >>= 1) { if (t < s) red[t] = fmaxf(red[t], red[t + s]); __syncthreads(); }
  mx = red[0]; __syncthreads();
  float sm = 0.f;
  for (int j = t; j < LQ; j += 256) sm += __expf(row[j] - mx);
  red[t] = sm; __syncthreads();
  for (int s = 128; s > 0; s >>= 1) { if (t < s) red[t] += red[t + s]; __syncthreads(); }
  if (t == 0) { rmax[blockIdx.x] = mx; rsum[blockIdx.x] = red[0]; }
}

__global__ __launch_bounds__(256) void k_colstats(const float* __restrict__ S,
                                                  float* cmax, float* csum) {
  __shared__ float red[256];
  int c = blockIdx.x, t = threadIdx.x;
  float mx = -3.4e38f;
  for (int j = t; j < LQ; j += 256) mx = fmaxf(mx, S[(size_t)j * LQ + c]);
  red[t] = mx; __syncthreads();
  for (int s = 128; s > 0; s >>= 1) { if (t < s) red[t] = fmaxf(red[t], red[t + s]); __syncthreads(); }
  mx = red[0]; __syncthreads();
  float sm = 0.f;
  for (int j = t; j < LQ; j += 256) sm += __expf(S[(size_t)j * LQ + c] - mx);
  red[t] = sm; __syncthreads();
  for (int s = 128; s > 0; s >>= 1) { if (t < s) red[t] += red[t + s]; __syncthreads(); }
  if (t == 0) { cmax[c] = mx; csum[c] = red[0]; }
}

__global__ void k_dualP(const float* __restrict__ S,
                        const float* __restrict__ rmax, const float* __restrict__ rsum,
                        const float* __restrict__ cmax, const float* __restrict__ csum,
                        float* __restrict__ P) {
  size_t i = (size_t)blockIdx.x * blockDim.x + threadIdx.x;
  if (i >= (size_t)LQ * LQ) return;
  int r = (int)(i / LQ), c = (int)(i % LQ);
  float s = S[i];
  P[i] = (__expf(s - rmax[r]) / rsum[r]) * (__expf(s - cmax[c]) / csum[c]);
}

__global__ __launch_bounds__(256) void k_rowargmax(const float* __restrict__ P,
                                                   int* __restrict__ idx, float* __restrict__ conf) {
  __shared__ float sv[256]; __shared__ int si[256];
  int r = blockIdx.x, t = threadIdx.x;
  const float* row = P + (size_t)r * LQ;
  float bv = -3.4e38f; int bi = 0;
  for (int j = t; j < LQ; j += 256) { float v = row[j]; if (v > bv) { bv = v; bi = j; } }
  sv[t] = bv; si[t] = bi; __syncthreads();
  for (int s = 128; s > 0; s >>= 1) {
    if (t < s && (sv[t + s] > sv[t] || (sv[t + s] == sv[t] && si[t + s] < si[t]))) {
      sv[t] = sv[t + s]; si[t] = si[t + s];
    }
    __syncthreads();
  }
  if (t == 0) { idx[r] = si[0]; if (conf) conf[r] = sv[0]; }
}

__global__ __launch_bounds__(256) void k_colargmax(const float* __restrict__ P,
                                                   int* __restrict__ idx) {
  __shared__ float sv[256]; __shared__ int si[256];
  int c = blockIdx.x, t = threadIdx.x;
  float bv = -3.4e38f; int bi = 0;
  for (int j = t; j < LQ; j += 256) { float v = P[(size_t)j * LQ + c]; if (v > bv) { bv = v; bi = j; } }
  sv[t] = bv; si[t] = bi; __syncthreads();
  for (int s = 128; s > 0; s >>= 1) {
    if (t < s && (sv[t + s] > sv[t] || (sv[t + s] == sv[t] && si[t + s] < si[t]))) {
      sv[t] = sv[t + s]; si[t] = si[t + s];
    }
    __syncthreads();
  }
  if (t == 0) idx[c] = si[0];
}

__global__ void k_match(const int* __restrict__ idx0, const int* __restrict__ idx1,
                        const float* __restrict__ conf,
                        float* __restrict__ valid, float* __restrict__ pt0,
                        int* mcx0, int* mcy0, int* mcx1, int* mcy1, int* minb,
                        float* b1x, float* b1y) {
  int l = blockIdx.x * blockDim.x + threadIdx.x;
  if (l >= LQ) return;
  int j = idx0[l];
  valid[l] = (idx1[j] == l && conf[l] > 0.2f) ? 1.f : 0.f;
  int m0x = l % W0Q, m0y = l / W0Q, m1x = j % W0Q, m1y = j / W0Q;
  pt0[2 * l]     = (float)(m0x * 8 + 4);
  pt0[2 * l + 1] = (float)(m0y * 8 + 4);
  int c0x = m0x * 4, c0y = m0y * 4, c1x = m1x * 4, c1y = m1y * 4;
  int inb = (c0x >= 2) && (c0x + 2 < WFQ) && (c0y >= 2) && (c0y + 2 < HFQ) &&
            (c1x >= 2) && (c1x + 2 < WFQ) && (c1y >= 2) && (c1y + 2 < HFQ);
  mcx0[l] = min(max(c0x, 2), WFQ - 3); mcy0[l] = min(max(c0y, 2), HFQ - 3);
  mcx1[l] = min(max(c1x, 2), WFQ - 3); mcy1[l] = min(max(c1y, 2), HFQ - 3);
  minb[l] = inb;
  b1x[l] = (float)(m1x * 8 + 4);
  b1y[l] = (float)(m1y * 8 + 4);
}

// per-match 5x5 correlation + softmax expectation; one wave per match
__global__ __launch_bounds__(32) void k_fine(const float* __restrict__ f0p,
                                             const float* __restrict__ f1p,
                                             const int* mcx0, const int* mcy0,
                                             const int* mcx1, const int* mcy1,
                                             const int* minb,
                                             const float* b1x, const float* b1y,
                                             float* __restrict__ pt1) {
  __shared__ float ctr[CFQ];
  __shared__ float cr[25];
  int l = blockIdx.x, t = threadIdx.x;
  int cx0 = mcx0[l], cy0 = mcy0[l], cx1 = mcx1[l], cy1 = mcy1[l];
  ctr[t] = f0p[(size_t)t * NPIX + cy0 * WFQ + cx0];
  __syncthreads();
  if (t < 25) {
    int dy = t / 5 - 2, dx = t % 5 - 2;
    int base = (cy1 + dy) * WFQ + (cx1 + dx);
    float s = 0.f;
#pragma unroll
    for (int c = 0; c < CFQ; ++c) s += ctr[c] * f1p[(size_t)c * NPIX + base];
    cr[t] = s;
  }
  __syncthreads();
  if (t == 0) {
    float mx = cr[0];
    for (int p = 1; p < 25; ++p) mx = fmaxf(mx, cr[p]);
    float sm = 0.f, ox = 0.f, oy = 0.f;
    for (int p = 0; p < 25; ++p) {
      float e = __expf(cr[p] - mx);
      sm += e; ox += e * (float)(p % 5 - 2); oy += e * (float)(p / 5 - 2);
    }
    ox /= sm; oy /= sm;
    float use = minb[l] ? 2.f : 0.f;   // scale = STRIDE // (STRIDE/2) = 2
    pt1[2 * l]     = b1x[l] + ox * use;
    pt1[2 * l + 1] = b1y[l] + oy * use;
  }
}

// ---------------------------------------------------------------------------
// host-side helpers
// ---------------------------------------------------------------------------
static inline void gemm(hipStream_t s, const float* A, int lda, const float* B, int ldb,
                        int bT, float* C, int ldc, const float* bias, int bm, int relu,
                        int M, int N, int K, float scale) {
  dim3 g(N / 32, M / 32);
  k_gemm<<<g, 32, 0, s>>>(A, lda, B, ldb, bT, C, ldc, bias, bm, relu, K, scale);
}

static void mha(hipStream_t s, const float* xq, const float* xkv,
                const float* inW, const float* inB, const float* outW, const float* outB,
                float* qkv, float* hcat, float* out) {
  if (xq == xkv) {
    gemm(s, xq, DIM, inW, DIM, 1, qkv, 3 * DIM, inB, 1, 0, LQ, 3 * DIM, DIM, 1.f);
  } else {
    gemm(s, xq,  DIM, inW,             DIM, 1, qkv,       3 * DIM, inB,       1, 0, LQ, DIM,     DIM, 1.f);
    gemm(s, xkv, DIM, inW + DIM * DIM, DIM, 1, qkv + DIM, 3 * DIM, inB + DIM, 1, 0, LQ, 2 * DIM, DIM, 1.f);
  }
  dim3 g(LQ / 16, NHEAD);
  k_flash<<<g, 32, 0, s>>>(qkv, hcat);
  gemm(s, hcat, DIM, outW, DIM, 1, out, DIM, outB, 1, 0, LQ, DIM, DIM, 1.f);
}

extern "C" void kernel_launch(void* const* d_in, const int* in_sizes, int n_in,
                              void* d_out, int out_size, void* d_ws, size_t ws_size,
                              hipStream_t stream) {
  (void)in_sizes; (void)n_in; (void)out_size; (void)ws_size;
  const float* coarse0  = (const float*)d_in[0];
  const float* coarse1  = (const float*)d_in[1];
  const float* fine0    = (const float*)d_in[2];
  const float* fine1    = (const float*)d_in[3];
  const float* sa_in_w  = (const float*)d_in[4];
  const float* sa_in_b  = (const float*)d_in[5];
  const float* sa_out_w = (const float*)d_in[6];
  const float* sa_out_b = (const float*)d_in[7];
  const float* ca_in_w  = (const float*)d_in[8];
  const float* ca_in_b  = (const float*)d_in[9];
  const float* ca_out_w = (const float*)d_in[10];
  const float* ca_out_b = (const float*)d_in[11];
  const float* sn_g = (const float*)d_in[12];
  const float* sn_b = (const float*)d_in[13];
  const float* cn_g = (const float*)d_in[14];
  const float* cn_b = (const float*)d_in[15];
  const float* ffn_w1 = (const float*)d_in[16];
  const float* ffn_b1 = (const float*)d_in[17];
  const float* ffn_w2 = (const float*)d_in[18];
  const float* ffn_b2 = (const float*)d_in[19];
  const float* fn_g = (const float*)d_in[20];
  const float* fn_b = (const float*)d_in[21];
  const float* proj_w = (const float*)d_in[22];
  const float* proj_b = (const float*)d_in[23];

  // workspace carve (floats, 64-element aligned)
  float* base = (float*)d_ws;
  size_t o = 0;
  auto A = [&](size_t n) { float* p = base + o; o += (n + 63) & ~(size_t)63; return p; };
  float* x0   = A((size_t)LQ * DIM);
  float* x1   = A((size_t)LQ * DIM);
  float* qkv0 = A((size_t)LQ * 3 * DIM);
  float* qkv1 = A((size_t)LQ * 3 * DIM);
  float* att0 = A((size_t)LQ * DIM);
  float* att1 = A((size_t)LQ * DIM);
  float* tmp0 = A((size_t)LQ * DIM);
  float* tmp1 = A((size_t)LQ * DIM);
  float* hid  = A((size_t)LQ * 2 * DIM);
  float* f0n  = A((size_t)LQ * DIM);
  float* f1n  = A((size_t)LQ * DIM);
  float* rmax = A(LQ); float* rsum = A(LQ);
  float* cmax = A(LQ); float* csum = A(LQ);
  float* f0p  = A((size_t)CFQ * NPIX);
  float* f1p  = A((size_t)CFQ * NPIX);
  float* b1x  = A(LQ); float* b1y = A(LQ);
  int* idx0 = (int*)A(LQ); int* idx1 = (int*)A(LQ);
  int* mcx0 = (int*)A(LQ); int* mcy0 = (int*)A(LQ);
  int* mcx1 = (int*)A(LQ); int* mcy1 = (int*)A(LQ);
  int* minb = (int*)A(LQ);
  float* Smat = A((size_t)LQ * LQ);   // similarity matrix only (attention fused)

  float* out   = (float*)d_out;
  float* P     = out;
  float* valid = out + (size_t)LQ * LQ;
  float* conf  = valid + LQ;
  float* pt0   = conf + LQ;
  float* pt1   = pt0 + 2 * LQ;

  // tokenize
  k_tok<<<(LQ * DIM + 255) / 256, 256, 0, stream>>>(coarse0, x0);
  k_tok<<<(LQ * DIM + 255) / 256, 256, 0, stream>>>(coarse1, x1);

  for (int i = 0; i < 2; ++i) {
    const float* saW  = sa_in_w  + (size_t)i * 3 * DIM * DIM;
    const float* saB  = sa_in_b  + (size_t)i * 3 * DIM;
    const float* saOW = sa_out_w + (size_t)i * DIM * DIM;
    const float* saOB = sa_out_b + (size_t)i * DIM;
    const float* caW  = ca_in_w  + (size_t)i * 3 * DIM * DIM;
    const float* caB  = ca_in_b  + (size_t)i * 3 * DIM;
    const float* caOW = ca_out_w + (size_t)i * DIM * DIM;
    const float* caOB = ca_out_b + (size_t)i * DIM;

    // self-attention (shared weights, independent streams)
    mha(stream, x0, x0, saW, saB, saOW, saOB, qkv0, att0, tmp0);
    k_add_ln<<<LQ, 128, 0, stream>>>(x0, tmp0, sn_g + i * DIM, sn_b + i * DIM);
    mha(stream, x1, x1, saW, saB, saOW, saOB, qkv1, att1, tmp1);
    k_add_ln<<<LQ, 128, 0, stream>>>(x1, tmp1, sn_g + i * DIM, sn_b + i * DIM);

    // cross-attention: both computed before residual/LN
    mha(stream, x0, x1, caW, caB, caOW, caOB, qkv0, att0, tmp0);
    mha(stream, x1, x0, caW, caB, caOW, caOB, qkv1, att1, tmp1);
    k_add_ln<<<LQ, 128, 0, stream>>>(x0, tmp0, cn_g + i * DIM, cn_b + i * DIM);
    k_add_ln<<<LQ, 128, 0, stream>>>(x1, tmp1, cn_g + i * DIM, cn_b + i * DIM);

    // FFN
    const float* w1 = ffn_w1 + (size_t)i * 2 * DIM * DIM;
    const float* b1 = ffn_b1 + (size_t)i * 2 * DIM;
    const float* w2 = ffn_w2 + (size_t)i * DIM * 2 * DIM;
    const float* b2 = ffn_b2 + (size_t)i * DIM;
    gemm(stream, x0, DIM, w1, DIM, 1, hid, 2 * DIM, b1, 1, 1, LQ, 2 * DIM, DIM, 1.f);
    gemm(stream, hid, 2 * DIM, w2, 2 * DIM, 1, tmp0, DIM, b2, 1, 0, LQ, DIM, 2 * DIM, 1.f);
    k_add_ln<<<LQ, 128, 0, stream>>>(x0, tmp0, fn_g + i * DIM, fn_b + i * DIM);
    gemm(stream, x1, DIM, w1, DIM, 1, hid, 2 * DIM, b1, 1, 1, LQ, 2 * DIM, DIM, 1.f);
    gemm(stream, hid, 2 * DIM, w2, 2 * DIM, 1, tmp1, DIM, b2, 1, 0, LQ, DIM, 2 * DIM, 1.f);
    k_add_ln<<<LQ, 128, 0, stream>>>(x1, tmp1, fn_g + i * DIM, fn_b + i * DIM);
  }

  // similarity + dual softmax -> P
  k_l2norm<<<LQ, 128, 0, stream>>>(x0, f0n);
  k_l2norm<<<LQ, 128, 0, stream>>>(x1, f1n);
  gemm(stream, f0n, DIM, f1n, DIM, 1, Smat, LQ, nullptr, 0, 0, LQ, LQ, DIM, 10.f); // 1/TEMP
  k_rowstats<<<LQ, 256, 0, stream>>>(Smat, rmax, rsum);
  k_colstats<<<LQ, 256, 0, stream>>>(Smat, cmax, csum);
  k_dualP<<<(int)(((size_t)LQ * LQ + 255) / 256), 256, 0, stream>>>(Smat, rmax, rsum, cmax, csum, P);

  // mutual-NN matching
  k_rowargmax<<<LQ, 256, 0, stream>>>(P, idx0, conf);
  k_colargmax<<<LQ, 256, 0, stream>>>(P, idx1);
  k_match<<<(LQ + 255) / 256, 256, 0, stream>>>(idx0, idx1, conf, valid, pt0,
                                                mcx0, mcy0, mcx1, mcy1, minb, b1x, b1y);

  // fine projection (32x32 GEMM over 33280 pixels, per-row bias) + refinement
  gemm(stream, proj_w, CFQ, fine0, NPIX, 0, f0p, NPIX, proj_b, 2, 0, CFQ, NPIX, CFQ, 1.f);
  gemm(stream, proj_w, CFQ, fine1, NPIX, 0, f1p, NPIX, proj_b, 2, 0, CFQ, NPIX, CFQ, 1.f);
  k_fine<<<LQ, 32, 0, stream>>>(f0p, f1p, mcx0, mcy0, mcx1, mcy1, minb, b1x, b1y, pt1);
}
